// AdvancedScorer_79345225826339
// MI455X (gfx1250) — compile-verified, matrix operand order checked by reading, not run on hardware
//
#include <hip/hip_runtime.h>
#include <hip/hip_bf16.h>

typedef __attribute__((ext_vector_type(16))) __bf16 v16bf;
typedef __attribute__((ext_vector_type(8)))  float  v8f;

#define EMBED_DIM 128
#define KDIM      256   // 2*EMBED_DIM
#define HID       512
#define LN_EPS    1e-5f

// ---------- helpers ----------
__device__ __forceinline__ unsigned short f2bf(float x) {
    unsigned u = __float_as_uint(x);
    unsigned r = (u + 0x7FFFu + ((u >> 16) & 1u)) >> 16;   // round-to-nearest-even
    return (unsigned short)r;
}

// float atomic max via signed/unsigned int ordering trick (works for mixed signs,
// seg_max initialized to -FLT_MAX). Lowers to global_atomic_max_i32 / min_u32.
__device__ __forceinline__ void atomicMaxF(float* addr, float v) {
    if (v >= 0.0f) atomicMax((int*)addr, __float_as_int(v));
    else           atomicMin((unsigned int*)addr, __float_as_uint(v));
}

// ---------- kernel 1: init accumulators ----------
__global__ __launch_bounds__(256) void init_kernel(float* __restrict__ seg_sum,
                                                   float* __restrict__ seg_max,
                                                   float* __restrict__ counts, int H) {
    int i = blockIdx.x * 256 + threadIdx.x;
    int total = H * EMBED_DIM;
    if (i < total) {
        seg_sum[i] = 0.0f;
        seg_max[i] = -3.402823466e38f;
    }
    if (i < H) counts[i] = 0.0f;
}

// ---------- kernel 2: pack W1 f32 -> bf16 in WMMA B-fragment order ----------
// Layout: w1p[(((kt*32 + ntile)*32 + lane)*16 + e], kt in [0,8), ntile in [0,32),
// lane in [0,32), e in [0,16). Per ISA 16-bit B 32x16 layout:
//   n = ntile*16 + (lane&15), half = lane>>4,
//   K = kt*32 + e + (e<8 ? 0 : 8) + half*8
__global__ __launch_bounds__(256) void pack_w1_kernel(const float* __restrict__ W1,
                                                      unsigned short* __restrict__ w1p) {
    int i = blockIdx.x * 256 + threadIdx.x;     // 0 .. 131071
    int e     = i & 15;
    int lane  = (i >> 4) & 31;
    int ntile = (i >> 9) & 31;
    int kt    = i >> 14;
    int half  = lane >> 4;
    int n     = ntile * 16 + (lane & 15);
    int K     = kt * 32 + e + ((e < 8) ? 0 : 8) + half * 8;
    w1p[i] = f2bf(W1[K * HID + n]);
}

// ---------- kernel 3: scatter (wave32 per incidence edge) ----------
__global__ __launch_bounds__(256) void scatter_kernel(const float* __restrict__ emb,
                                                      const int*   __restrict__ nidx,
                                                      const int*   __restrict__ eidx,
                                                      float* __restrict__ seg_sum,
                                                      float* __restrict__ seg_max,
                                                      float* __restrict__ counts, int E) {
    int wid  = (int)((blockIdx.x * 256 + threadIdx.x) >> 5);
    int lane = threadIdx.x & 31;
    if (wid >= E) return;
    int n = nidx[wid];
    int e = eidx[wid];
    const float4 v = *(const float4*)(emb + (size_t)n * EMBED_DIM + lane * 4);
    float* ps = seg_sum + (size_t)e * EMBED_DIM + lane * 4;
    float* pm = seg_max + (size_t)e * EMBED_DIM + lane * 4;
    atomicAdd(ps + 0, v.x); atomicAdd(ps + 1, v.y);
    atomicAdd(ps + 2, v.z); atomicAdd(ps + 3, v.w);
    atomicMaxF(pm + 0, v.x); atomicMaxF(pm + 1, v.y);
    atomicMaxF(pm + 2, v.z); atomicMaxF(pm + 3, v.w);
    if (lane == 0) atomicAdd(counts + e, 1.0f);
}

// ---------- kernel 4: finalize mean/max -> bf16 h rows [Hp, 256] ----------
__global__ __launch_bounds__(256) void finalize_kernel(const float* __restrict__ seg_sum,
                                                       const float* __restrict__ seg_max,
                                                       const float* __restrict__ counts,
                                                       unsigned short* __restrict__ hbf,
                                                       int H, int Hp) {
    int i = blockIdx.x * 256 + threadIdx.x;
    if (i >= Hp * EMBED_DIM) return;
    int h = i >> 7;            // /128
    int d = i & 127;
    size_t base = (size_t)h * KDIM + d;
    if (h >= H) {              // zero pad rows (feeds WMMA harmlessly)
        hbf[base] = 0; hbf[base + EMBED_DIM] = 0;
        return;
    }
    float c    = counts[h];
    float mean = seg_sum[i] / fmaxf(c, 1.0f);
    float mx   = (c > 0.0f) ? seg_max[i] : 0.0f;
    hbf[base]             = f2bf(mean);
    hbf[base + EMBED_DIM] = f2bf(mx);
}

// ---------- kernel 5: fused GEMM(bf16 WMMA) + bias + LayerNorm + ReLU + GEMV ----------
// Block: 256 threads = 8 waves. Computes 16 rows x 512 cols.
__global__ __launch_bounds__(256) void mlp_kernel(const unsigned short* __restrict__ hbf,
                                                  const unsigned short* __restrict__ w1p,
                                                  const float* __restrict__ b1,
                                                  const float* __restrict__ gamma,
                                                  const float* __restrict__ beta,
                                                  const float* __restrict__ W2,
                                                  const float* __restrict__ b2,
                                                  float* __restrict__ out, int H) {
    __shared__ unsigned short hA[16 * KDIM];   // 8 KB  A tile (bf16)
    __shared__ float          zb[16 * HID];    // 32 KB z tile (f32)

    const int tid  = threadIdx.x;
    const int row0 = blockIdx.x * 16;

    // Stage A tile: 8192 B via two b128 per thread
    {
        const uint4* src = (const uint4*)(hbf + (size_t)row0 * KDIM);
        uint4* dst = (uint4*)hA;
        dst[tid]       = src[tid];
        dst[tid + 256] = src[tid + 256];
    }
    __syncthreads();

    const int wave = tid >> 5;
    const int lane = tid & 31;
    const int half = lane >> 4;
    const int m    = lane & 15;

    v8f acc[4];
    #pragma unroll
    for (int j = 0; j < 4; ++j) acc[j] = (v8f){0,0,0,0,0,0,0,0};

    union UA { v16bf v; unsigned u[8]; };

    #pragma unroll
    for (int kt = 0; kt < 8; ++kt) {
        // A fragment from LDS (16-bit A 16x32 layout; contiguous K pairs -> ds_load_b32)
        UA a;
        #pragma unroll
        for (int v = 0; v < 8; ++v) {
            int K0 = 2 * v + ((v >= 4) ? 8 : 0) + half * 8;
            a.u[v] = *(const unsigned*)&hA[m * KDIM + kt * 32 + K0];
        }
        // 4 N-tiles per wave; B fragments pre-swizzled -> two global_load_b128 each
        #pragma unroll
        for (int j = 0; j < 4; ++j) {
            int ntile = wave * 4 + j;
            const v16bf b = *(const v16bf*)(w1p + ((((kt * 32 + ntile) * 32) + lane) << 4));
            acc[j] = __builtin_amdgcn_wmma_f32_16x16x32_bf16(
                false, a.v, false, b, (short)0, acc[j], false, false);
        }
    }

    // C layout: VGPR r, lane l -> M = r + 8*(l>>4), N = ntile*16 + (l&15)
    #pragma unroll
    for (int j = 0; j < 4; ++j) {
        int n = (wave * 4 + j) * 16 + m;
        #pragma unroll
        for (int r = 0; r < 8; ++r) {
            int mm = r + half * 8;
            zb[mm * HID + n] = acc[j][r] + b1[n];
        }
    }
    __syncthreads();

    // LayerNorm + ReLU + dot(W2): 16 threads per row, strided cols (bank-friendly)
    const int row = tid >> 4;
    const int sub = tid & 15;
    float s = 0.0f, sq = 0.0f;
    const float* zr = &zb[row * HID];
    #pragma unroll
    for (int c = 0; c < 32; ++c) {
        float x = zr[sub + 16 * c];
        s += x; sq += x * x;
    }
    #pragma unroll
    for (int msk = 1; msk < 16; msk <<= 1) {
        s  += __shfl_xor(s,  msk, 32);
        sq += __shfl_xor(sq, msk, 32);
    }
    float mean = s * (1.0f / HID);
    float var  = sq * (1.0f / HID) - mean * mean;
    float rstd = rsqrtf(var + LN_EPS);

    float dot = 0.0f;
    #pragma unroll
    for (int c = 0; c < 32; ++c) {
        int col = sub + 16 * c;
        float x = (zr[col] - mean) * rstd * gamma[col] + beta[col];
        x = fmaxf(x, 0.0f);
        dot += x * W2[col];
    }
    #pragma unroll
    for (int msk = 1; msk < 16; msk <<= 1) dot += __shfl_xor(dot, msk, 32);

    int gr = row0 + row;
    if (sub == 0 && gr < H) out[gr] = dot + b2[0];
}

// ---------- launch ----------
extern "C" void kernel_launch(void* const* d_in, const int* in_sizes, int n_in,
                              void* d_out, int out_size, void* d_ws, size_t ws_size,
                              hipStream_t stream) {
    const float* node_embeds = (const float*)d_in[0];
    const int*   hyperedge   = (const int*)d_in[1];     // [2, E] stacked
    const float* W1          = (const float*)d_in[2];   // [256, 512]
    const float* b1          = (const float*)d_in[3];
    const float* gamma       = (const float*)d_in[4];
    const float* beta        = (const float*)d_in[5];
    const float* W2          = (const float*)d_in[6];   // [512]
    const float* b2          = (const float*)d_in[7];
    float*       out         = (float*)d_out;

    const int E  = in_sizes[1] / 2;
    const int H  = out_size;
    const int Hp = ((H + 15) / 16) * 16;

    const int* node_idx = hyperedge;
    const int* edge_idx = hyperedge + E;

    // workspace carve-up (256B aligned)
    char* ws = (char*)d_ws;
    size_t o = 0;
    auto align256 = [](size_t x) { return (x + 255) & ~(size_t)255; };
    float* seg_sum = (float*)(ws + o);          o = align256(o + (size_t)H * EMBED_DIM * 4);
    float* seg_max = (float*)(ws + o);          o = align256(o + (size_t)H * EMBED_DIM * 4);
    float* counts  = (float*)(ws + o);          o = align256(o + (size_t)H * 4);
    unsigned short* hbf = (unsigned short*)(ws + o); o = align256(o + (size_t)Hp * KDIM * 2);
    unsigned short* w1p = (unsigned short*)(ws + o); o = align256(o + (size_t)KDIM * HID * 2);
    (void)ws_size; (void)n_in;

    // 1) init accumulators
    {
        int total = H * EMBED_DIM;
        init_kernel<<<(total + 255) / 256, 256, 0, stream>>>(seg_sum, seg_max, counts, H);
    }
    // 2) pack W1 -> bf16 B-fragments
    pack_w1_kernel<<<(KDIM * HID) / 256, 256, 0, stream>>>(W1, w1p);
    // 3) scatter: one wave32 per incidence edge
    scatter_kernel<<<(E + 7) / 8, 256, 0, stream>>>(node_embeds, node_idx, edge_idx,
                                                    seg_sum, seg_max, counts, E);
    // 4) finalize mean/max -> bf16 h (with zero padding rows)
    finalize_kernel<<<(Hp * EMBED_DIM + 255) / 256, 256, 0, stream>>>(seg_sum, seg_max, counts,
                                                                      hbf, H, Hp);
    // 5) fused WMMA GEMM + LN + ReLU + GEMV
    mlp_kernel<<<Hp / 16, 256, 0, stream>>>(hbf, w1p, b1, gamma, beta, W2, b2, out, H);
}
